// GraphVAE_10952166604982
// MI455X (gfx1250) — matrix operand order repaired and checked by reading, not compiled, not run on hardware
//
#include <hip/hip_runtime.h>
#include <math.h>

// ---------------------------------------------------------------------------
// GraphVAE forward for MI455X (gfx1250, wave32).
// All GEMMs run through v_wmma_f32_16x16x32_bf16 (bf16 inputs, f32 accum).
// Block tile 128x64, wave tile 32x32 (2x2 WMMA accums), K step 64
// => 8 WMMAs per barrier per wave. Global->LDS staging uses the CDNA5
// async-DMA path (global_load_async_to_lds_b128) with DOUBLE-BUFFERED LDS:
// stage t+1 streams in (s_wait_asynccnt 6 = oldest-6-done) while stage t
// feeds the WMMA pipe, so DMA latency overlaps matrix math.
// ---------------------------------------------------------------------------

typedef unsigned int u32;
typedef __attribute__((ext_vector_type(16))) __bf16 v16bf;
typedef __attribute__((ext_vector_type(8)))  float  v8f;
typedef __attribute__((ext_vector_type(4)))  u32    u32x4;

static constexpr int BATCH   = 4096;
static constexpr int D_NODES = 16;
static constexpr int D_LAT   = 64;
static constexpr int E_HID   = 512;
static constexpr int E_OUT   = 256;
static constexpr int DEC_HID = 512;
static constexpr int X_DIM   = 3 * 32 * 32;   // 3072

// ---- helpers ---------------------------------------------------------------

__device__ __forceinline__ unsigned short f2bf(float f) {
    union { float f; u32 u; } x; x.f = f;
    u32 u = x.u;
    return (unsigned short)((u + 0x7FFFu + ((u >> 16) & 1u)) >> 16);  // RNE
}

__device__ __forceinline__ u32 hashu(u32 x) {
    x ^= x >> 16; x *= 0x7feb352du;
    x ^= x >> 15; x *= 0x846ca68bu;
    x ^= x >> 16; return x;
}

__device__ __forceinline__ float rng_uniform(u32 seed, u32 idx) {
    u32 a = hashu(seed * 0x9E3779B9u ^ hashu(idx));
    return ((a >> 8) + 1u) * (1.0f / 16777218.0f);
}

__device__ __forceinline__ float rng_normal(u32 seed, u32 idx) {
    u32 a = hashu(seed * 0x9E3779B9u ^ hashu(idx));
    u32 b = hashu(a ^ 0x85ebca6bu);
    float u1 = ((a >> 8) + 1u) * (1.0f / 16777218.0f);
    float u2 = (b >> 8) * (1.0f / 16777216.0f);
    return sqrtf(-2.0f * logf(u1)) * __cosf(6.28318530718f * u2);
}

// Async 16-byte global -> LDS copy (per lane), tracked by ASYNCcnt.
__device__ __forceinline__ void async_copy_b128(const unsigned short* gptr,
                                                const unsigned short* lptr) {
    u32 lds_off = (u32)(size_t)lptr;   // flat addr[31:0] == LDS byte offset
    asm volatile("global_load_async_to_lds_b128 %0, %1, off"
                 :: "v"(lds_off), "v"(gptr) : "memory");
}

// ---- generic bf16 WMMA GEMM ------------------------------------------------
// C[M,N] = act(A[M,K] @ B[K,N] + bias).
// Grid: (N/64, M/128, batches), 256 threads = 8 waves (4 x 2 wave grid).
// flags: bit0 = ReLU, bit1 = store bf16 (else f32).

union FragU { u32 u[8]; u32x4 q[2]; v16bf v; };

static constexpr int LDA_S = 72;   // 64 + 8 pad (bf16 elements)
static constexpr int LDB_S = 72;   // 64 + 8 pad

// Issue one stage of async tile DMA: 6 x b128 per thread (A 128x64, B 64x64).
__device__ __forceinline__ void stage_tiles(
    const unsigned short* Ab, const unsigned short* Bb,
    unsigned short* la, unsigned short* lb,
    int m0, int n0, int k0, int lda, int ldb, int tid)
{
    #pragma unroll
    for (int c = 0; c < 4; ++c) {               // A: 128x64 bf16
        int id = tid + 256 * c;
        int row = id >> 3, cg = (id & 7) * 8;
        async_copy_b128(Ab + (m0 + row) * lda + k0 + cg, la + row * LDA_S + cg);
    }
    #pragma unroll
    for (int c = 0; c < 2; ++c) {               // B: 64x64 bf16
        int id = tid + 256 * c;
        int row = id >> 3, cg = (id & 7) * 8;
        async_copy_b128(Bb + (k0 + row) * ldb + n0 + cg, lb + row * LDB_S + cg);
    }
}

__global__ __launch_bounds__(256) void gemm_bf16_wmma(
    const unsigned short* __restrict__ A, const unsigned short* __restrict__ Bm,
    const float* __restrict__ bias, void* __restrict__ Cout,
    int M, int N, int K, int lda, int ldb, int ldc,
    long long bsA, long long bsB, long long bsC, long long bsBias, int flags)
{
    __shared__ unsigned short lA[2][128 * LDA_S];   // double-buffered A tile
    __shared__ unsigned short lB[2][64  * LDB_S];   // double-buffered B tile

    const int tid  = threadIdx.x;
    const int lane = tid & 31;
    const int wave = tid >> 5;          // 8 waves
    const int wm   = wave >> 1;         // 0..3  -> 32-row slice
    const int wn   = wave & 1;          // 0..1  -> 32-col slice
    const int m0   = blockIdx.y * 128;
    const int n0   = blockIdx.x * 64;
    const int bz   = blockIdx.z;

    const unsigned short* Ab = A  + (long long)bz * bsA;
    const unsigned short* Bb = Bm + (long long)bz * bsB;

    const int li = lane & 15, half = lane >> 4;

    v8f acc[2][2] = {{{}, {}}, {{}, {}}};

    // Prologue: start DMA for stage 0.
    stage_tiles(Ab, Bb, lA[0], lB[0], m0, n0, 0, lda, ldb, tid);
    int cur = 0;

    for (int k0 = 0; k0 < K; k0 += 64) {
        if (k0 + 64 < K) {
            // Kick off stage t+1 into the other buffer, then require only the
            // OLDEST 6 async ops (stage t) to be done: DMA overlaps compute.
            stage_tiles(Ab, Bb, lA[cur ^ 1], lB[cur ^ 1],
                        m0, n0, k0 + 64, lda, ldb, tid);
            asm volatile("s_wait_asynccnt 0x6" ::: "memory");
        } else {
            asm volatile("s_wait_asynccnt 0x0" ::: "memory");
        }
        __syncthreads();

        const unsigned short* la = lA[cur];
        const unsigned short* lb = lB[cur];

        // ---- two 32-deep WMMA slabs per stage ----
        #pragma unroll
        for (int ks = 0; ks < 64; ks += 32) {
            FragU fa[2], fb[2];
            // A fragment (16x32, lane=M): two contiguous 8-elem K runs per lane
            #pragma unroll
            for (int bm = 0; bm < 2; ++bm) {
                const unsigned short* pa =
                    &la[(wm * 32 + bm * 16 + li) * LDA_S + ks + half * 8];
                fa[bm].q[0] = *(const u32x4*)pa;        // K = ks + half*8 .. +7
                fa[bm].q[1] = *(const u32x4*)(pa + 16); // K = ks + 16 + half*8 ..
            }
            // B fragment (32x16, lane=N): K in same two 8-elem runs as A
            #pragma unroll
            for (int bn = 0; bn < 2; ++bn) {
                const unsigned short* pb =
                    &lb[ks * LDB_S + wn * 32 + bn * 16 + li];
                #pragma unroll
                for (int j = 0; j < 8; ++j) {
                    int kk = (j < 4) ? (half * 8 + 2 * j) : (16 + half * 8 + 2 * j - 8);
                    fb[bn].u[j] = (u32)pb[kk * LDB_S] | ((u32)pb[(kk + 1) * LDB_S] << 16);
                }
            }
            #pragma unroll
            for (int bm = 0; bm < 2; ++bm)
                #pragma unroll
                for (int bn = 0; bn < 2; ++bn)
                    acc[bm][bn] = __builtin_amdgcn_wmma_f32_16x16x32_bf16(
                        false, fa[bm].v, false, fb[bn].v, (short)0,
                        acc[bm][bn], false, false);
        }
        __syncthreads();   // protect buf[cur] before next iteration's DMA
        cur ^= 1;
    }

    // ---- epilogue: C VGPR r -> row (half*8 + r), col = lane&15 ----
    const float* biasb = bias ? (bias + (long long)bz * bsBias) : nullptr;
    #pragma unroll
    for (int bm = 0; bm < 2; ++bm) {
        #pragma unroll
        for (int bn = 0; bn < 2; ++bn) {
            #pragma unroll
            for (int r = 0; r < 8; ++r) {
                int row = m0 + wm * 32 + bm * 16 + half * 8 + r;
                int col = n0 + wn * 32 + bn * 16 + li;
                float v = acc[bm][bn][r];
                if (biasb) v += biasb[col];
                if (flags & 1) v = fmaxf(v, 0.0f);
                long long cidx = (long long)bz * bsC + (long long)(row * ldc + col);
                if (flags & 2) ((unsigned short*)Cout)[cidx] = f2bf(v);
                else           ((float*)Cout)[cidx] = v;
            }
        }
    }
}

// ---- elementwise kernels ---------------------------------------------------

__global__ void cvt_bf16_kernel(const float* __restrict__ in,
                                unsigned short* __restrict__ out, long long n) {
    long long i = (long long)blockIdx.x * blockDim.x + threadIdx.x;
    if (i < n) out[i] = f2bf(in[i]);
}

__global__ void init_state_kernel(float* pm, float* pv, float* qm, float* qv,
                                  long long n) {
    long long i = (long long)blockIdx.x * blockDim.x + threadIdx.x;
    if (i < n) { pm[i] = 0.0f; pv[i] = 1.0f; qm[i] = 0.0f; qv[i] = 1.0f; }
}

// Build z_parents[b, j*64+t] = gate_j(b) * (prior_mu + sqrt(prior_var)*eps),
// gates = hard Gumbel one-hot (forward value of hard gumbel-softmax).
__global__ void topdown_prepare_kernel(
    const float* __restrict__ prior_mu, const float* __restrict__ prior_var,
    const float* __restrict__ gating, unsigned short* __restrict__ zpar,
    int i, int n_anc)
{
    long long idx = (long long)blockIdx.x * blockDim.x + threadIdx.x;
    long long total = (long long)BATCH * n_anc * 64;
    if (idx >= total) return;
    int t = (int)(idx & 63);
    int j = (int)((idx >> 6) % n_anc);
    int b = (int)(idx / ((long long)64 * n_anc));

    int best = 0; float bestv = -1e30f;
    for (int jj = 0; jj < n_anc; ++jj) {
        float u = rng_uniform(2u * i + 1u, (u32)(b * (D_NODES - 1) + jj));
        float g = -logf(-logf(u));
        float v = gating[i * (D_NODES - 1) + jj] + g;   // tau>0 monotone: argmax only
        if (v > bestv) { bestv = v; best = jj; }
    }
    float gate = (j == best) ? 1.0f : 0.0f;

    int node = i + 1 + j;
    long long pidx = ((long long)b * D_NODES + node) * D_LAT + t;
    float eps = rng_normal(2u * i, (u32)(((long long)b * n_anc + j) * 64 + t));
    float anc = prior_mu[pidx] + sqrtf(prior_var[pidx]) * eps;
    zpar[(long long)b * (n_anc * 64) + j * 64 + t] = f2bf(gate * anc);
}

// Precision-weighted posterior fusion + prior update for node i.
__global__ void topdown_epilogue_kernel(
    const float* __restrict__ tdout, const float* __restrict__ bu,
    float* __restrict__ prior_mu, float* __restrict__ prior_var,
    float* __restrict__ post_mu, float* __restrict__ post_var, int i)
{
    long long idx = (long long)blockIdx.x * blockDim.x + threadIdx.x;
    if (idx >= (long long)BATCH * D_LAT) return;
    int t = (int)(idx & 63);
    int b = (int)(idx >> 6);

    float mu_hat     = tdout[(long long)b * 128 + t];
    float logvar_hat = tdout[(long long)b * 128 + 64 + t];
    float var_hat    = __expf(logvar_hat);

    long long pidx = ((long long)b * D_NODES + i) * D_LAT + t;
    prior_mu[pidx]  = mu_hat;
    prior_var[pidx] = var_hat;

    long long buidx = ((long long)b * D_NODES + i) * 128 + t;
    float mu_t = bu[buidx];
    float v1   = __expf(bu[buidx + 64]);
    float sigma = 1.0f / (1.0f / v1 + 1.0f / var_hat);
    float mu    = sigma * (mu_t / v1) + mu_hat / var_hat;   // (faithful to reference)
    post_mu[pidx]  = mu;
    post_var[pidx] = sigma;
}

__global__ void sample_z_kernel(const float* __restrict__ post_mu,
                                const float* __restrict__ post_var,
                                unsigned short* __restrict__ zbf) {
    long long idx = (long long)blockIdx.x * blockDim.x + threadIdx.x;
    if (idx >= (long long)BATCH * D_NODES * D_LAT) return;
    float eps = rng_normal(1000u, (u32)idx);
    float z = post_mu[idx] + sqrtf(post_var[idx]) * eps;
    zbf[idx] = f2bf(z);
}

__global__ void final_sample_kernel(const float* __restrict__ dec,
                                    float* __restrict__ out) {
    long long idx = (long long)blockIdx.x * blockDim.x + threadIdx.x;
    if (idx >= (long long)BATCH * X_DIM) return;
    int b = (int)(idx / X_DIM);
    int x = (int)(idx % X_DIM);
    float mu = dec[(long long)b * (2 * X_DIM) + x];
    float lv = dec[(long long)b * (2 * X_DIM) + X_DIM + x];
    float eps = rng_normal(1001u, (u32)idx);
    out[idx] = mu + __expf(0.5f * lv) * eps;
}

// ---- host driver -----------------------------------------------------------

extern "C" void kernel_launch(void* const* d_in, const int* in_sizes, int n_in,
                              void* d_out, int out_size, void* d_ws, size_t ws_size,
                              hipStream_t stream) {
    const float* img    = (const float*)d_in[0];
    const float* enc_w1 = (const float*)d_in[1];
    const float* enc_b1 = (const float*)d_in[2];
    const float* enc_w2 = (const float*)d_in[3];
    const float* enc_b2 = (const float*)d_in[4];
    const float* bu_w   = (const float*)d_in[5];
    const float* bu_b   = (const float*)d_in[6];
    const float* gating = (const float*)d_in[7];
    const float* td_w   = (const float*)d_in[8];
    const float* td_b   = (const float*)d_in[9];
    const float* dec_w1 = (const float*)d_in[10];
    const float* dec_b1 = (const float*)d_in[11];
    const float* dec_w2 = (const float*)d_in[12];
    const float* dec_b2 = (const float*)d_in[13];

    char* ws = (char*)d_ws;
    auto up = [](size_t x) { return (x + 255) & ~(size_t)255; };

    // Region 0: early bf16 activations, later aliased by the f32 decoder output.
    size_t o = 0;
    unsigned short* Xbf    = (unsigned short*)(ws + o); o += up((size_t)BATCH * X_DIM * 2);
    unsigned short* h1bf   = (unsigned short*)(ws + o); o += up((size_t)BATCH * E_HID * 2);
    unsigned short* embbf  = (unsigned short*)(ws + o); o += up((size_t)BATCH * E_OUT * 2);
    unsigned short* zparbf = (unsigned short*)(ws + o); o += up((size_t)BATCH * (D_NODES - 1) * D_LAT * 2);
    unsigned short* zbf    = (unsigned short*)(ws + o); o += up((size_t)BATCH * D_NODES * D_LAT * 2);
    float* decout = (float*)(ws + 0);                  // aliases region 0 (dead by then)
    size_t region0 = up((size_t)BATCH * 2 * X_DIM * 4);
    o = (o > region0) ? o : region0;

    // Weights (bf16), live across the whole pass.
    unsigned short* W1bf  = (unsigned short*)(ws + o); o += up((size_t)X_DIM * E_HID * 2);
    unsigned short* W2bf  = (unsigned short*)(ws + o); o += up((size_t)E_HID * E_OUT * 2);
    unsigned short* buwbf = (unsigned short*)(ws + o); o += up((size_t)D_NODES * E_OUT * 2 * D_LAT * 2);
    unsigned short* tdwbf = (unsigned short*)(ws + o); o += up((size_t)D_NODES * (D_LAT * (D_NODES - 1)) * 2 * D_LAT * 2);
    unsigned short* dw1bf = (unsigned short*)(ws + o); o += up((size_t)D_NODES * D_LAT * DEC_HID * 2);
    unsigned short* dw2bf = (unsigned short*)(ws + o); o += up((size_t)DEC_HID * 2 * X_DIM * 2);

    // f32 state, live across the top-down chain.
    float* buF     = (float*)(ws + o); o += up((size_t)BATCH * D_NODES * 2 * D_LAT * 4);
    float* priorMu = (float*)(ws + o); o += up((size_t)BATCH * D_NODES * D_LAT * 4);
    float* priorVa = (float*)(ws + o); o += up((size_t)BATCH * D_NODES * D_LAT * 4);
    float* postMu  = (float*)(ws + o); o += up((size_t)BATCH * D_NODES * D_LAT * 4);
    float* postVa  = (float*)(ws + o); o += up((size_t)BATCH * D_NODES * D_LAT * 4);
    float* tdout   = (float*)(ws + o); o += up((size_t)BATCH * 2 * D_LAT * 4);
    unsigned short* hdecbf = (unsigned short*)(ws + o); o += up((size_t)BATCH * DEC_HID * 2);

    auto cvt = [&](const float* src, unsigned short* dst, long long n) {
        cvt_bf16_kernel<<<dim3((unsigned)((n + 255) / 256)), 256, 0, stream>>>(src, dst, n);
    };
    auto gemm = [&](const unsigned short* A, const unsigned short* Bm, const float* bias,
                    void* C, int M, int N, int K, int lda, int ldb, int ldc,
                    long long bsA, long long bsB, long long bsC, long long bsBias,
                    int batches, int flags) {
        dim3 grid(N / 64, M / 128, batches);
        gemm_bf16_wmma<<<grid, 256, 0, stream>>>(A, Bm, bias, C, M, N, K,
                                                 lda, ldb, ldc, bsA, bsB, bsC, bsBias, flags);
    };

    // --- stage inputs as bf16 ---
    cvt(img,    Xbf,   (long long)BATCH * X_DIM);
    cvt(enc_w1, W1bf,  (long long)X_DIM * E_HID);
    cvt(enc_w2, W2bf,  (long long)E_HID * E_OUT);
    cvt(bu_w,   buwbf, (long long)D_NODES * E_OUT * 2 * D_LAT);
    cvt(td_w,   tdwbf, (long long)D_NODES * (D_LAT * (D_NODES - 1)) * 2 * D_LAT);
    cvt(dec_w1, dw1bf, (long long)D_NODES * D_LAT * DEC_HID);
    cvt(dec_w2, dw2bf, (long long)DEC_HID * 2 * X_DIM);

    long long nstate = (long long)BATCH * D_NODES * D_LAT;
    init_state_kernel<<<dim3((unsigned)((nstate + 255) / 256)), 256, 0, stream>>>(
        priorMu, priorVa, postMu, postVa, nstate);

    // --- encoder ---
    gemm(Xbf,  W1bf, enc_b1, h1bf,  BATCH, E_HID, X_DIM, X_DIM, E_HID, E_HID,
         0, 0, 0, 0, 1, /*relu|bf16*/ 3);
    gemm(h1bf, W2bf, enc_b2, embbf, BATCH, E_OUT, E_HID, E_HID, E_OUT, E_OUT,
         0, 0, 0, 0, 1, /*bf16*/ 2);

    // --- bottom-up projection: 16 batched GEMMs, C interleaved [B,16,128] ---
    gemm(embbf, buwbf, bu_b, buF, BATCH, 2 * D_LAT, E_OUT,
         E_OUT, 2 * D_LAT, D_NODES * 2 * D_LAT,
         0, (long long)E_OUT * 2 * D_LAT, 2 * D_LAT, 2 * D_LAT,
         D_NODES, /*f32*/ 0);

    // --- sequential top-down chain ---
    for (int i = D_NODES - 2; i >= 0; --i) {
        int n_anc = (D_NODES - 1) - i;
        long long np = (long long)BATCH * n_anc * 64;
        topdown_prepare_kernel<<<dim3((unsigned)((np + 255) / 256)), 256, 0, stream>>>(
            priorMu, priorVa, gating, zparbf, i, n_anc);

        gemm(zparbf, tdwbf + (long long)i * (D_LAT * (D_NODES - 1)) * 2 * D_LAT,
             td_b + (long long)i * 2 * D_LAT, tdout,
             BATCH, 2 * D_LAT, n_anc * 64, n_anc * 64, 2 * D_LAT, 2 * D_LAT,
             0, 0, 0, 0, 1, /*f32*/ 0);

        long long ne = (long long)BATCH * D_LAT;
        topdown_epilogue_kernel<<<dim3((unsigned)((ne + 255) / 256)), 256, 0, stream>>>(
            tdout, buF, priorMu, priorVa, postMu, postVa, i);
    }

    // --- sample z, decode ---
    sample_z_kernel<<<dim3((unsigned)((nstate + 255) / 256)), 256, 0, stream>>>(
        postMu, postVa, zbf);

    gemm(zbf,    dw1bf, dec_b1, hdecbf, BATCH, DEC_HID, D_NODES * D_LAT,
         D_NODES * D_LAT, DEC_HID, DEC_HID, 0, 0, 0, 0, 1, /*relu|bf16*/ 3);
    gemm(hdecbf, dw2bf, dec_b2, decout, BATCH, 2 * X_DIM, DEC_HID,
         DEC_HID, 2 * X_DIM, 2 * X_DIM, 0, 0, 0, 0, 1, /*f32*/ 0);

    long long nout = (long long)BATCH * X_DIM;
    final_sample_kernel<<<dim3((unsigned)((nout + 255) / 256)), 256, 0, stream>>>(
        decout, (float*)d_out);
}
